// MambaSLRUL_73443940761736
// MI455X (gfx1250) — matrix operand order repaired and checked by reading, not compiled, not run on hardware
//
#include <hip/hip_runtime.h>
#include <hip/hip_bf16.h>
#include <math.h>
#include <stdint.h>

// Problem constants (from reference)
#define BQ     16
#define LQ     4096
#define CINQ   32
#define DMQ    256
#define DSQ    16
#define DCONVQ 4
#define DIQ    512      // EXP * DM
#define DTRQ   16       // DM / 16
#define KCONV  3

typedef __bf16 bf16_t;
typedef bf16_t v16bf __attribute__((ext_vector_type(16)));
typedef float  v8f   __attribute__((ext_vector_type(8)));

typedef unsigned int u32x4 __attribute__((ext_vector_type(4)));
typedef int          i32x8 __attribute__((ext_vector_type(8)));
typedef int          i32x4 __attribute__((ext_vector_type(4)));

#if defined(__has_builtin)
#  if __has_builtin(__builtin_amdgcn_tensor_load_to_lds) && \
      __has_builtin(__builtin_amdgcn_s_wait_tensorcnt)
#    define HAVE_TDM 1
#  endif
#endif
#ifndef HAVE_TDM
#  define HAVE_TDM 0
#endif

#if HAVE_TDM
// ---------------------------------------------------------------------------
// Tensor Data Mover: linear 1-D tile load Global -> LDS (CDNA5 ISA ch.8 D#).
// D# group0: count=1 | lds_addr | global_addr[56:0] | type=2.
// D# group1: data_size code, tensor_dim0 = tile_dim0 = nelem, tensor_dim1 = 1,
//            dim0 stride = nelem; padding/iterate/multicast disabled.
// Groups 2/3 zero (<=2D tensor); trailing 256-bit group unused (zero).
// Issued once per wave; tracked by TENSORcnt (s_wait_tensorcnt).
// dsz_code: 0=1B, 1=2B, 2=4B, 3=8B elements. nelem must fit in 16 bits.
// This toolchain exposes the 6-arg builtin:
//   (u32x4 g0, i32x8 g1, i32x4 g2, i32x4 g3, i32x8 g4, i32 cpol)
// ---------------------------------------------------------------------------
__device__ __forceinline__ void tdm_load_linear(uint32_t lds_off, uint64_t gaddr,
                                                uint32_t nelem, uint32_t dsz_code) {
  u32x4 g0;
  g0[0] = 1u;                                                   // count=1, user D#
  g0[1] = lds_off;                                              // lds_addr
  g0[2] = (uint32_t)gaddr;                                      // global_addr[31:0]
  g0[3] = (uint32_t)((gaddr >> 32) & 0x01FFFFFFu) | (2u << 30); // [56:32] | type=2
  i32x8 g1;
  g1[0] = (int)(dsz_code << 16);                // wg_mask=0 | data_size
  g1[1] = (int)(nelem << 16);                   // tensor_dim0[15:0]
  g1[2] = (int)((nelem >> 16) | (1u << 16));    // tensor_dim0[31:16] | tensor_dim1=1
  g1[3] = (int)((nelem & 0xFFFFu) << 16);       // tile_dim0
  g1[4] = 0;                                    // tile_dim1/2 unused
  g1[5] = (int)nelem;                           // tensor_dim0_stride lo
  g1[6] = 0;
  g1[7] = 0;
  i32x4 g2 = {0, 0, 0, 0};
  i32x4 g3 = {0, 0, 0, 0};
  i32x8 g4 = {0, 0, 0, 0, 0, 0, 0, 0};
  __builtin_amdgcn_tensor_load_to_lds(g0, g1, g2, g3, g4, 0);
}
#endif

// ---------------------------------------------------------------------------
// WMMA tile loaders per CDNA5 ISA §7.12.2 (wave32).
// A (16x32 bf16): lane l -> row m=l&15; e=0..7 -> K=kb+e, e=8..15 -> K=kb+16+(e-8),
//                 kb = 32*kk + 8*(l>=16).
// B (32x16 bf16): lane l -> col n=l&15; same K mapping as A (kb = 32*kk + 8*(l>=16)).
// C/D (16x16 f32): lane l -> col n=l&15; vgpr r -> row m = r + 8*(l>=16).
// ---------------------------------------------------------------------------
__device__ __forceinline__ v16bf wmma_load_a(const bf16_t* base, int lda, int kk, int lane) {
  const int m  = lane & 15;
  const int hi = (lane >> 4) & 1;
  const bf16_t* p = base + (size_t)m * lda + kk * 32 + hi * 8;
  v16bf a;
#pragma unroll
  for (int e = 0; e < 8; ++e) { a[e] = p[e]; a[e + 8] = p[16 + e]; }
  return a;
}

__device__ __forceinline__ v16bf wmma_load_b(const bf16_t* W, int ldw, int ntile, int kk, int lane) {
  const int n  = lane & 15;
  const int hi = (lane >> 4) & 1;
  const bf16_t* p = W + (size_t)(ntile * 16 + n) * ldw + kk * 32 + hi * 8;
  v16bf b;
#pragma unroll
  for (int e = 0; e < 8; ++e) { b[e] = p[e]; b[e + 8] = p[16 + e]; }
  return b;
}

__device__ __forceinline__ float silu_f(float v) { return v / (1.f + __expf(-v)); }

// ---------------------------------------------------------------------------
// K0: convert big weight matrices to bf16 row-major copies; zero pooled accum.
// ---------------------------------------------------------------------------
__global__ void k0_prep(const float* __restrict__ w1, const float* __restrict__ wx,
                        const float* __restrict__ w4,
                        bf16_t* __restrict__ W1, bf16_t* __restrict__ Wx,
                        bf16_t* __restrict__ W4, float* __restrict__ pooled) {
  const size_t n1 = (size_t)2 * DIQ * DMQ;   // 1024 x 256
  const size_t nx = (size_t)48 * DIQ;        // 48 x 512
  const size_t n4 = (size_t)DMQ * DIQ;       // 256 x 512
  const size_t np = (size_t)BQ * DMQ;
  const size_t stride = (size_t)gridDim.x * blockDim.x;
  const size_t t0 = (size_t)blockIdx.x * blockDim.x + threadIdx.x;
  for (size_t i = t0; i < n1; i += stride) W1[i] = (bf16_t)w1[i];
  for (size_t i = t0; i < nx; i += stride) Wx[i] = (bf16_t)wx[i];
  for (size_t i = t0; i < n4; i += stride) W4[i] = (bf16_t)w4[i];
  for (size_t i = t0; i < np; i += stride) pooled[i] = 0.f;
}

// ---------------------------------------------------------------------------
// K1: fused edge-pad conv(CIN=32,K=3) + sinusoidal PE -> 16x256 bf16 LDS tile,
//     then in_proj GEMM (16x1024 = emb @ W1^T) via bf16 WMMA; writes xz bf16.
// ---------------------------------------------------------------------------
__global__ __launch_bounds__(128) void k1_emb_inproj(
    const float* __restrict__ x, const float* __restrict__ conv_w,
    const bf16_t* __restrict__ W1, const float* __restrict__ in_b,
    bf16_t* __restrict__ xz) {
  __shared__ bf16_t smE[16 * DMQ];
  const int b  = blockIdx.y;
  const int l0 = blockIdx.x * 16;
  const int tid = threadIdx.x;

  for (int i = tid; i < 16 * DMQ; i += 128) {
    const int m = i >> 8;
    const int c = i & (DMQ - 1);
    const int l = l0 + m;
    float acc = 0.f;
#pragma unroll
    for (int w = 0; w < KCONV; ++w) {
      int lj = l - 1 + w;
      lj = lj < 0 ? 0 : (lj > LQ - 1 ? LQ - 1 : lj);   // edge padding
      const float* xr = x + ((size_t)b * LQ + lj) * CINQ;
      const float* wr = conv_w + (size_t)c * CINQ * KCONV + w;
#pragma unroll
      for (int ci = 0; ci < CINQ; ++ci) acc += xr[ci] * wr[ci * KCONV];
    }
    const float div = __expf((float)(c & ~1) * (-9.210340371976184f / (float)DMQ));
    const float ang = (float)l * div;
    acc += (c & 1) ? __cosf(ang) : __sinf(ang);
    smE[i] = (bf16_t)acc;
  }
  __syncthreads();

  const int lane = tid & 31;
  const int wv   = tid >> 5;
  const int n    = lane & 15;
  const int mbase = (lane & 16) ? 8 : 0;

  v16bf areg[8];
#pragma unroll
  for (int kk = 0; kk < 8; ++kk) areg[kk] = wmma_load_a(smE, DMQ, kk, lane);

  for (int nt = wv; nt < 64; nt += 4) {
    if (nt + 4 < 64)
      __builtin_prefetch(W1 + (size_t)((nt + 4) * 16 + n) * DMQ, 0, 1);
    v8f acc = {};
#pragma unroll
    for (int kk = 0; kk < 8; ++kk) {
      v16bf bm = wmma_load_b(W1, DMQ, nt, kk, lane);
      acc = __builtin_amdgcn_wmma_f32_16x16x32_bf16(false, areg[kk], false, bm,
                                                    (short)0, acc, false, false);
    }
    const float bias = in_b[nt * 16 + n];
#pragma unroll
    for (int r = 0; r < 8; ++r) {
      const int m = mbase + r;
      xz[((size_t)b * LQ + (l0 + m)) * (size_t)(2 * DIQ) + nt * 16 + n] =
          (bf16_t)(acc[r] + bias);
    }
  }
}

// ---------------------------------------------------------------------------
// K2: depthwise causal conv1d (DCONV=4, zero left pad) + bias + SiLU -> xc bf16
// ---------------------------------------------------------------------------
__global__ __launch_bounds__(256) void k2_dwconv(
    const bf16_t* __restrict__ xz, const float* __restrict__ cw,
    const float* __restrict__ cb, bf16_t* __restrict__ xc) {
  const size_t idx = (size_t)blockIdx.x * 256 + threadIdx.x;
  if (idx >= (size_t)BQ * LQ * DIQ) return;
  const int d = (int)(idx & (DIQ - 1));
  const size_t bl = idx >> 9;              // b*L + l
  const int l = (int)(bl & (LQ - 1));
  float s = cb[d];
#pragma unroll
  for (int j = 0; j < DCONVQ; ++j) {
    const int lj = l - (DCONVQ - 1) + j;
    if (lj >= 0)
      s += (float)xz[(bl - (DCONVQ - 1) + j) * (size_t)(2 * DIQ) + d] * cw[d * DCONVQ + j];
  }
  xc[idx] = (bf16_t)silu_f(s);
}

// ---------------------------------------------------------------------------
// K3: x_proj GEMM: dbl(16x48) = xc(16x512) @ Wx^T, bf16 WMMA, fp32 out.
// 96 threads = 3 wave32, one N-tile each. xc tile staged via TDM (linear span).
// ---------------------------------------------------------------------------
__global__ __launch_bounds__(96) void k3_xproj(
    const bf16_t* __restrict__ xc, const bf16_t* __restrict__ Wx,
    float* __restrict__ dbl) {
  __shared__ bf16_t smA[16 * DIQ];
  const int tile = blockIdx.x;
  const int b  = tile / (LQ / 16);
  const int l0 = (tile % (LQ / 16)) * 16;
  const int tid = threadIdx.x;
  const bf16_t* src = xc + ((size_t)b * LQ + l0) * DIQ;
#if HAVE_TDM
  if (tid < 32) {  // wave 0 issues the DMA, waits, barrier publishes
    tdm_load_linear((uint32_t)(uintptr_t)smA, (uint64_t)(uintptr_t)src,
                    16 * DIQ, /*2B*/ 1);
    __builtin_amdgcn_s_wait_tensorcnt(0);
  }
#else
  for (int i = tid; i < 16 * DIQ; i += 96) smA[i] = src[i];
#endif
  __syncthreads();

  const int lane = tid & 31;
  const int wv   = tid >> 5;     // n-tile 0..2
  v8f acc = {};
#pragma unroll
  for (int kk = 0; kk < 16; ++kk) {
    v16bf a  = wmma_load_a(smA, DIQ, kk, lane);
    v16bf bm = wmma_load_b(Wx, DIQ, wv, kk, lane);
    acc = __builtin_amdgcn_wmma_f32_16x16x32_bf16(false, a, false, bm,
                                                  (short)0, acc, false, false);
  }
  const int n = lane & 15;
  const int mbase = (lane & 16) ? 8 : 0;
#pragma unroll
  for (int r = 0; r < 8; ++r)
    dbl[((size_t)b * LQ + (l0 + mbase + r)) * 48 + wv * 16 + n] = acc[r];
}

// ---------------------------------------------------------------------------
// K4: fused selective scan; one thread per channel d. dt_proj + softplus +
// recurrence (DS=16 states in registers) + D*xc skip + silu(z) gating.
// dbl rows double-buffered through LDS: wave0 issues TDM for chunk t+1,
// computes chunk t, then s_wait_tensorcnt + barrier. y overwrites xc in place.
// ---------------------------------------------------------------------------
__global__ __launch_bounds__(128) void k4_scan(
    const float* __restrict__ dbl, bf16_t* __restrict__ xc_y,
    const bf16_t* __restrict__ xz, const float* __restrict__ dtw_g,
    const float* __restrict__ dtb_g, const float* __restrict__ Alog,
    const float* __restrict__ Dg) {
  __shared__ float smD[2][32 * 48];
  const int b = blockIdx.y;
  const int d = blockIdx.x * 128 + threadIdx.x;

  float dtw[DTRQ], Aa[DSQ], h[DSQ];
#pragma unroll
  for (int s = 0; s < DTRQ; ++s) dtw[s] = dtw_g[d * DTRQ + s];
#pragma unroll
  for (int s = 0; s < DSQ; ++s) { Aa[s] = -__expf(Alog[d * DSQ + s]); h[s] = 0.f; }
  const float dtb = dtb_g[d];
  const float Dd  = Dg[d];
  const size_t rowbase = (size_t)b * LQ;

#if HAVE_TDM
  if (threadIdx.x < 32) {
    tdm_load_linear((uint32_t)(uintptr_t)&smD[0][0],
                    (uint64_t)(uintptr_t)(dbl + rowbase * 48), 32 * 48, /*4B*/ 2);
    __builtin_amdgcn_s_wait_tensorcnt(0);
  }
  __syncthreads();
#endif

  int buf = 0;
  for (int t0 = 0; t0 < LQ; t0 += 32) {
#if HAVE_TDM
    if (t0 + 32 < LQ && threadIdx.x < 32)   // prefetch next chunk (async)
      tdm_load_linear((uint32_t)(uintptr_t)&smD[buf ^ 1][0],
                      (uint64_t)(uintptr_t)(dbl + (rowbase + t0 + 32) * 48),
                      32 * 48, 2);
#else
    __syncthreads();
    const float* gsrc = dbl + (rowbase + t0) * 48;
    for (int i = threadIdx.x; i < 32 * 48; i += 128) smD[buf][i] = gsrc[i];
    __syncthreads();
#endif
#pragma unroll 2
    for (int ti = 0; ti < 32; ++ti) {
      const int t = t0 + ti;
      const float* row = smD[buf] + ti * 48;
      float dt = dtb;
#pragma unroll
      for (int s = 0; s < DTRQ; ++s) dt += row[s] * dtw[s];
      dt = (dt > 20.f) ? dt : log1pf(__expf(dt));        // softplus
      const size_t off = (rowbase + t) * (size_t)DIQ + d;
      const float xv  = (float)xc_y[off];
      const float dtx = dt * xv;
      float acc = 0.f;
#pragma unroll
      for (int s = 0; s < DSQ; ++s) {
        const float dA = __expf(dt * Aa[s]);
        h[s] = dA * h[s] + dtx * row[16 + s];
        acc += h[s] * row[32 + s];
      }
      const float z = (float)xz[(rowbase + t) * (size_t)(2 * DIQ) + DIQ + d];
      xc_y[off] = (bf16_t)((acc + Dd * xv) * silu_f(z));
    }
#if HAVE_TDM
    if (threadIdx.x < 32) __builtin_amdgcn_s_wait_tensorcnt(0);
    __syncthreads();
    buf ^= 1;
#endif
  }
}

// ---------------------------------------------------------------------------
// K5: out_proj GEMM (16x256 = y(16x512) @ W4^T) via bf16 WMMA (A-chunks
// preloaded once, reused over 4 N-tiles/wave), then fused LayerNorm + SiLU +
// (1/L)-mean pooling via LDS + global f32 atomics. y tile staged via TDM.
// ---------------------------------------------------------------------------
__global__ __launch_bounds__(128) void k5_outproj_ln(
    const bf16_t* __restrict__ y, const bf16_t* __restrict__ W4,
    const float* __restrict__ ln_g, const float* __restrict__ ln_b,
    float* __restrict__ pooled) {
  __shared__ bf16_t smY[16 * DIQ];
  __shared__ float  smOut[16 * DMQ];
  __shared__ float  smPool[DMQ];
  const int tile = blockIdx.x;
  const int b  = tile / (LQ / 16);
  const int l0 = (tile % (LQ / 16)) * 16;
  const int tid = threadIdx.x;
  const bf16_t* src = y + ((size_t)b * LQ + l0) * DIQ;
#if HAVE_TDM
  if (tid < 32) {
    tdm_load_linear((uint32_t)(uintptr_t)smY, (uint64_t)(uintptr_t)src,
                    16 * DIQ, /*2B*/ 1);
    __builtin_amdgcn_s_wait_tensorcnt(0);
  }
#else
  for (int i = tid; i < 16 * DIQ; i += 128) smY[i] = src[i];
#endif
  for (int c = tid; c < DMQ; c += 128) smPool[c] = 0.f;
  __syncthreads();

  const int lane = tid & 31;
  const int wv   = tid >> 5;
  const int n    = lane & 15;
  const int mbase = (lane & 16) ? 8 : 0;

  v16bf areg[16];
#pragma unroll
  for (int kk = 0; kk < 16; ++kk) areg[kk] = wmma_load_a(smY, DIQ, kk, lane);

  for (int nt = wv; nt < 16; nt += 4) {
    v8f acc = {};
#pragma unroll
    for (int kk = 0; kk < 16; ++kk) {
      v16bf bm = wmma_load_b(W4, DIQ, nt, kk, lane);
      acc = __builtin_amdgcn_wmma_f32_16x16x32_bf16(false, areg[kk], false, bm,
                                                    (short)0, acc, false, false);
    }
#pragma unroll
    for (int r = 0; r < 8; ++r)
      smOut[(mbase + r) * DMQ + nt * 16 + n] = acc[r];
  }
  __syncthreads();

  // Wave wv handles rows 4*wv..4*wv+3; full-wave32 shuffle reduction per row.
  for (int rr = 0; rr < 4; ++rr) {
    const int m = wv * 4 + rr;
    const float* row = smOut + m * DMQ;
    float s = 0.f, s2 = 0.f;
#pragma unroll
    for (int e = 0; e < 8; ++e) { const float v = row[lane * 8 + e]; s += v; s2 += v * v; }
#pragma unroll
    for (int off = 16; off > 0; off >>= 1) {
      s  += __shfl_xor(s, off, 32);
      s2 += __shfl_xor(s2, off, 32);
    }
    const float mu   = s * (1.f / DMQ);
    const float var  = s2 * (1.f / DMQ) - mu * mu;
    const float rstd = rsqrtf(var + 1e-5f);
#pragma unroll
    for (int e = 0; e < 8; ++e) {
      const int c = lane * 8 + e;
      float hv = (row[c] - mu) * rstd * ln_g[c] + ln_b[c];
      atomicAdd(&smPool[c], silu_f(hv));
    }
  }
  __syncthreads();
  const float invL = 1.f / (float)LQ;
  for (int c = tid; c < DMQ; c += 128)
    atomicAdd(&pooled[b * DMQ + c], smPool[c] * invL);
}

// ---------------------------------------------------------------------------
// K6: head: out[b] = pooled[b,:] . head_w + head_b   (one wave32 per batch)
// ---------------------------------------------------------------------------
__global__ __launch_bounds__(32) void k6_head(
    const float* __restrict__ pooled, const float* __restrict__ hw,
    const float* __restrict__ hb, float* __restrict__ out) {
  const int b = blockIdx.x;
  const int lane = threadIdx.x;
  float s = 0.f;
#pragma unroll
  for (int e = 0; e < 8; ++e) {
    const int c = lane * 8 + e;
    s += pooled[b * DMQ + c] * hw[c];
  }
#pragma unroll
  for (int off = 16; off > 0; off >>= 1) s += __shfl_xor(s, off, 32);
  if (lane == 0) out[b] = s + hb[0];
}

// ---------------------------------------------------------------------------
extern "C" void kernel_launch(void* const* d_in, const int* in_sizes, int n_in,
                              void* d_out, int out_size, void* d_ws, size_t ws_size,
                              hipStream_t stream) {
  const float* x         = (const float*)d_in[0];
  const float* conv_w    = (const float*)d_in[1];
  const float* in_proj_w = (const float*)d_in[2];
  const float* in_proj_b = (const float*)d_in[3];
  const float* conv1d_w  = (const float*)d_in[4];
  const float* conv1d_b  = (const float*)d_in[5];
  const float* x_proj_w  = (const float*)d_in[6];
  const float* dt_proj_w = (const float*)d_in[7];
  const float* dt_proj_b = (const float*)d_in[8];
  const float* A_log     = (const float*)d_in[9];
  const float* Dvec      = (const float*)d_in[10];
  const float* out_proj_w= (const float*)d_in[11];
  const float* ln_g      = (const float*)d_in[12];
  const float* ln_b      = (const float*)d_in[13];
  const float* head_w    = (const float*)d_in[14];
  const float* head_b    = (const float*)d_in[15];
  float* out = (float*)d_out;

  // Workspace layout (bytes)
  char* ws = (char*)d_ws;
  const size_t szW1  = (size_t)2 * DIQ * DMQ * sizeof(bf16_t);     // 512 KB
  const size_t szWx  = (size_t)48 * DIQ * sizeof(bf16_t);          // 48 KB
  const size_t szW4  = (size_t)DMQ * DIQ * sizeof(bf16_t);         // 256 KB
  const size_t szPool= (size_t)BQ * DMQ * sizeof(float);           // 16 KB
  const size_t szXZ  = (size_t)BQ * LQ * 2 * DIQ * sizeof(bf16_t); // 128 MB
  const size_t szXC  = (size_t)BQ * LQ * DIQ * sizeof(bf16_t);     // 64 MB

  bf16_t* W1  = (bf16_t*)ws;
  bf16_t* Wx  = (bf16_t*)(ws + szW1);
  bf16_t* W4  = (bf16_t*)(ws + szW1 + szWx);
  float* pooled = (float*)(ws + szW1 + szWx + szW4);
  bf16_t* xz  = (bf16_t*)(ws + szW1 + szWx + szW4 + szPool);
  bf16_t* xc  = (bf16_t*)((char*)xz + szXZ);                       // y in place
  float*  dbl = (float*)((char*)xc + szXC);                        // (B,L,48) f32

  k0_prep<<<512, 256, 0, stream>>>(in_proj_w, x_proj_w, out_proj_w, W1, Wx, W4, pooled);
  k1_emb_inproj<<<dim3(LQ / 16, BQ), 128, 0, stream>>>(x, conv_w, W1, in_proj_b, xz);
  k2_dwconv<<<(int)(((size_t)BQ * LQ * DIQ) / 256), 256, 0, stream>>>(xz, conv1d_w, conv1d_b, xc);
  k3_xproj<<<BQ * LQ / 16, 96, 0, stream>>>(xc, Wx, dbl);
  k4_scan<<<dim3(DIQ / 128, BQ), 128, 0, stream>>>(dbl, xc, xz, dt_proj_w, dt_proj_b, A_log, Dvec);
  k5_outproj_ln<<<BQ * LQ / 16, 128, 0, stream>>>(xc, W4, ln_g, ln_b, pooled);
  k6_head<<<BQ, 32, 0, stream>>>(pooled, head_w, head_b, out);

  (void)in_sizes; (void)n_in; (void)out_size; (void)ws_size;
}
